// context2query_attention_17858474017440
// MI455X (gfx1250) — matrix operand order repaired and checked by reading, not compiled, not run on hardware
//
#include <hip/hip_runtime.h>

#define Bn   64
#define LCn  1024
#define LQn  128
#define Dn   128
#define CTn  128
#define PITCH 136
#define NEGV 1e12f

typedef __bf16 v16bf __attribute__((ext_vector_type(16)));
typedef __bf16 bf8v  __attribute__((ext_vector_type(8)));
typedef float  v8f   __attribute__((ext_vector_type(8)));

// ---- WMMA fragment builders (CDNA5 16x16x32 bf16 layouts, wave32) ----
// Both builders issue exactly two aligned 16-byte LDS vector loads.

__device__ __forceinline__ v16bf frag_a_rowmajor(const __bf16* base, int m0, int k0, int lane) {
  const int m = lane & 15, hi = lane >> 4;
  const __bf16* p = base + (m0 + m) * PITCH + k0 + hi * 8;
  bf8v lo = *(const bf8v*)(p);
  bf8v h8 = *(const bf8v*)(p + 16);
  return __builtin_shufflevector(lo, h8, 0,1,2,3,4,5,6,7,8,9,10,11,12,13,14,15);
}

__device__ __forceinline__ v16bf frag_b_from_rows(const __bf16* base, int n0, int k0, int lane) {
  const int n = lane & 15, hi = lane >> 4;
  const __bf16* p = base + (n0 + n) * PITCH + k0 + hi * 8;
  bf8v lo = *(const bf8v*)(p);
  bf8v h8 = *(const bf8v*)(p + 16);
  return __builtin_shufflevector(lo, h8, 0,1,2,3,4,5,6,7,8,9,10,11,12,13,14,15);
}

__device__ __forceinline__ v8f wmma_bf16(v16bf a, v16bf b, v8f c) {
  return __builtin_amdgcn_wmma_f32_16x16x32_bf16(false, a, false, b, (short)0, c, false, false);
}

// ======================= Kernel 1: column softmax -> T =======================
// Grid Bn*2: each WG handles one batch and one q-half (64 q's).
// T[q,d] = sum_c softmax_c(S)[c,q] * xc[c,d]  (exact per-q; no cross-WG data).
// Also writes the out[:, :, 0:128] = x_cont passthrough (q-half 0 only).
__global__ __launch_bounds__(256)
void k1_col_softmax_T(const float* __restrict__ xc, const float* __restrict__ xq,
                      const int* __restrict__ cont_len,
                      const float* __restrict__ W0, const float* __restrict__ W1,
                      const float* __restrict__ W2, const float* __restrict__ bias,
                      __bf16* __restrict__ Tn, float* __restrict__ out) {
  __shared__ __align__(16) __bf16 s_xq[64 * PITCH];    // xq rows q0..q0+63, [q][d]
  __shared__ __align__(16) __bf16 s_xcw[CTn * PITCH];  // (xc*W2) [c][d]; reused as P^T [q][c]
  __shared__ __align__(16) __bf16 s_xcT[Dn * PITCH];   // xc transposed [d][c]; reused as f32 partial-T
  __shared__ float s_r0[CTn], s_q1[64], s_colsum[64], s_w0[Dn], s_w2[Dn];

  __bf16* s_pT = s_xcw;            // alias (dead after S-WMMA)
  float* s_red = (float*)s_xcT;    // alias (dead after last T-WMMA): 64x128 f32

  const int b = blockIdx.x >> 1;
  const int qh = blockIdx.x & 1;
  const int q0 = qh * 64;
  const int tid = threadIdx.x;
  const int lane = tid & 31;
  const int wv = tid >> 5;
  const int m0 = wv * 16;          // c-row block for S phase
  const int qt = wv >> 1;          // local q-tile (0..3) for T phase
  const int kh = wv & 1;           // K-half for T phase

  const float* xqb = xq + (size_t)b * LQn * Dn;
  const float* xcb = xc + (size_t)b * LCn * Dn;

  if (tid < 128) { s_w0[tid] = W0[tid]; s_w2[tid] = W2[tid]; }
  if (tid < 64) {
    s_colsum[tid] = 0.f;
    float a1 = 0.f;
    const float* rp = xqb + (size_t)(q0 + tid) * Dn;
    for (int d = 0; d < Dn; ++d) a1 += rp[d] * W1[d];
    s_q1[tid] = a1 + bias[q0 + tid];
  }
  for (int i = tid; i < 64 * Dn; i += 256) {
    int r = i >> 7, c = i & 127;
    s_xq[r * PITCH + c] = (__bf16)xqb[(size_t)(q0 + r) * Dn + c];
  }
  __syncthreads();

  const int cl = cont_len[b];

  v8f tacc[8];
#pragma unroll
  for (int j = 0; j < 8; ++j)
#pragma unroll
    for (int e = 0; e < 8; ++e) tacc[j][e] = 0.f;

  for (int ct = 0; ct < 8; ++ct) {
    const int c0 = ct * CTn;
    for (int i = tid; i < CTn * Dn; i += 256) {
      int r = i >> 7, c = i & 127;
      float v = xcb[(size_t)(c0 + r) * Dn + c];
      s_xcw[r * PITCH + c] = (__bf16)(v * s_w2[c]);
      s_xcT[c * PITCH + r] = (__bf16)v;
      if (qh == 0) out[((size_t)b * LCn + c0 + r) * (4 * Dn) + c] = v;  // passthrough
    }
    if (tid < 128) {
      const float* rp = xcb + (size_t)(c0 + tid) * Dn;
      float a0 = 0.f;
      for (int d = 0; d < Dn; ++d) a0 += rp[d] * s_w0[d];
      s_r0[tid] = a0;
    }
    if (ct < 7) {  // prefetch next xc tile (2 x 128B lines per thread = 64KB)
      const float* nf = xcb + (size_t)(c0 + CTn) * Dn;
      __builtin_prefetch(nf + tid * 32, 0, 0);
      __builtin_prefetch(nf + 8192 + tid * 32, 0, 0);
    }
    __syncthreads();

    // S half-tile = (xc*W2) @ xq^T : rows m0..m0+15, 64 q columns
    v8f sacc[4];
#pragma unroll
    for (int j = 0; j < 4; ++j)
#pragma unroll
      for (int e = 0; e < 8; ++e) sacc[j][e] = 0.f;
#pragma unroll
    for (int kk = 0; kk < 4; ++kk) {
      v16bf a = frag_a_rowmajor(s_xcw, m0, kk * 32, lane);
      v16bf bb[4];
#pragma unroll
      for (int j = 0; j < 4; ++j) bb[j] = frag_b_from_rows(s_xq, j * 16, kk * 32, lane);
#pragma unroll
      for (int j = 0; j < 4; ++j) sacc[j] = wmma_bf16(a, bb[j], sacc[j]);
    }
    __syncthreads();  // all waves done reading s_xcw/s_xq before P^T overwrite

    // add row/col terms, cont-mask, exp -> P^T [q][c] (packed 16B stores)
    {
      const int hi = lane >> 4, nl = lane & 15;
#pragma unroll
      for (int j = 0; j < 4; ++j) {
        const int ncol = j * 16 + nl;  // local q
        bf8v pv;
#pragma unroll
        for (int r = 0; r < 8; ++r) {
          const int mrow = m0 + 8 * hi + r;  // c within tile
          float v = sacc[j][r] + s_r0[mrow] + s_q1[ncol];
          if (c0 + mrow >= cl) v -= NEGV;  // column-softmax mask on c
          pv[r] = (__bf16)__expf(v);
        }
        *(bf8v*)(s_pT + ncol * PITCH + m0 + 8 * hi) = pv;
      }
    }
    __syncthreads();

    // column sums: row q of P^T is contiguous
    if (tid < 64) {
      float cs = 0.f;
      for (int c = 0; c < CTn; ++c) cs += (float)s_pT[tid * PITCH + c];
      s_colsum[tid] += cs;
    }

    // T += P^T @ xc : wave pair (qt, kh); each wave does 2 of 4 K-steps
#pragma unroll
    for (int kx = 0; kx < 2; ++kx) {
      const int kk = kh * 2 + kx;
      v16bf a = frag_a_rowmajor(s_pT, qt * 16, kk * 32, lane);
      v16bf bb[8];
#pragma unroll
      for (int j = 0; j < 8; ++j) bb[j] = frag_b_from_rows(s_xcT, j * 16, kk * 32, lane);
#pragma unroll
      for (int j = 0; j < 8; ++j) tacc[j] = wmma_bf16(a, bb[j], tacc[j]);
    }
    __syncthreads();
  }

  // merge K-half partials via LDS, normalize, write T (bf16) to workspace
  {
    const int hi = lane >> 4, nl = lane & 15;
    if (kh == 1) {
#pragma unroll
      for (int j = 0; j < 8; ++j)
#pragma unroll
        for (int r = 0; r < 8; ++r)
          s_red[(qt * 16 + r + 8 * hi) * Dn + j * 16 + nl] = tacc[j][r];
    }
    __syncthreads();
    if (kh == 0) {
      __bf16* tb = Tn + (size_t)b * LQn * Dn;
#pragma unroll
      for (int j = 0; j < 8; ++j) {
#pragma unroll
        for (int r = 0; r < 8; ++r) {
          const int mq = qt * 16 + r + 8 * hi;  // local q
          const int nd = j * 16 + nl;
          float v = tacc[j][r] + s_red[mq * Dn + nd];
          tb[(size_t)(q0 + mq) * Dn + nd] = (__bf16)(v / s_colsum[mq]);
        }
      }
    }
  }
}

// =========== Kernel 2: row softmax, c2q, q2c, fused concat output ===========
__global__ __launch_bounds__(256)
void k2_row_softmax_out(const float* __restrict__ xc, const float* __restrict__ xq,
                        const int* __restrict__ ques_len,
                        const float* __restrict__ W0, const float* __restrict__ W1,
                        const float* __restrict__ W2, const float* __restrict__ bias,
                        const __bf16* __restrict__ Tn, float* __restrict__ out) {
  __shared__ __align__(16) __bf16 s_xq[LQn * PITCH];   // xq [q][d]; reused as T^T [d][q]
  __shared__ __align__(16) __bf16 s_xqT[Dn * PITCH];   // xq transposed [d][q]
  __shared__ __align__(16) __bf16 s_xcw[CTn * PITCH];  // (xc*W2) [c][d]; reused as P [c][q]
  __shared__ float s_r0[CTn], s_q1[LQn], s_rowsum[CTn];

  __bf16* s_p = s_xcw;   // alias
  __bf16* s_tnT = s_xq;  // alias

  const int b = blockIdx.x >> 3;
  const int ct = blockIdx.x & 7;
  const int c0 = ct * CTn;
  const int tid = threadIdx.x;
  const int lane = tid & 31;
  const int wv = tid >> 5;
  const int m0 = wv * 16;

  const float* xqb = xq + (size_t)b * LQn * Dn;
  const float* xcb = xc + (size_t)b * LCn * Dn;
  const __bf16* tb = Tn + (size_t)b * LQn * Dn;

  if (tid < 128) {
    float a1 = 0.f;
    const float* rp = xqb + (size_t)tid * Dn;
    for (int d = 0; d < Dn; ++d) a1 += rp[d] * W1[d];
    s_q1[tid] = a1 + bias[tid];
  }
  for (int i = tid; i < LQn * Dn; i += 256) {
    int r = i >> 7, c = i & 127;
    __bf16 v = (__bf16)xqb[i];
    s_xq[r * PITCH + c] = v;
    s_xqT[c * PITCH + r] = v;
  }
  for (int i = tid; i < CTn * Dn; i += 256) {
    int r = i >> 7, c = i & 127;
    float v = xcb[(size_t)(c0 + r) * Dn + c];
    s_xcw[r * PITCH + c] = (__bf16)(v * W2[c]);
  }
  if (tid < 128) {
    const float* rp = xcb + (size_t)(c0 + tid) * Dn;
    float a0 = 0.f;
    for (int d = 0; d < Dn; ++d) a0 += rp[d] * W0[d];
    s_r0[tid] = a0;
  }
  __syncthreads();

  const int ql = ques_len[b];

  // S tile
  v8f sacc[8];
#pragma unroll
  for (int j = 0; j < 8; ++j)
#pragma unroll
    for (int e = 0; e < 8; ++e) sacc[j][e] = 0.f;
#pragma unroll
  for (int kk = 0; kk < 4; ++kk) {
    v16bf a = frag_a_rowmajor(s_xcw, m0, kk * 32, lane);
    v16bf bb[8];
#pragma unroll
    for (int j = 0; j < 8; ++j) bb[j] = frag_b_from_rows(s_xq, j * 16, kk * 32, lane);
#pragma unroll
    for (int j = 0; j < 8; ++j) sacc[j] = wmma_bf16(a, bb[j], sacc[j]);
  }
  __syncthreads();  // all waves done reading s_xcw/s_xq before overwrite

  // extras + ques mask + exp -> P [c][q]; row sums via xor-shuffles (16-lane half)
  {
    const int hi = lane >> 4, nl = lane & 15;
#pragma unroll
    for (int r = 0; r < 8; ++r) {
      const int mrow = m0 + r + 8 * hi;
      float rs = 0.f;
#pragma unroll
      for (int j = 0; j < 8; ++j) {
        int ncol = j * 16 + nl;
        float v = sacc[j][r] + s_r0[mrow] + s_q1[ncol];
        if (ncol >= ql) v -= NEGV;  // row-softmax mask on q
        float p = __expf(v);
        s_p[mrow * PITCH + ncol] = (__bf16)p;
        rs += p;
      }
      rs += __shfl_xor(rs, 1, 32);
      rs += __shfl_xor(rs, 2, 32);
      rs += __shfl_xor(rs, 4, 32);
      rs += __shfl_xor(rs, 8, 32);
      if (nl == 0) s_rowsum[mrow] = rs;
    }
  }
  // stage T^T [d][q] into the (dead) row-major xq buffer; packed 2-elem loads
  for (int i = tid; i < (LQn * Dn) / 2; i += 256) {
    union { unsigned u; __bf16 h[2]; } cv;
    cv.u = ((const unsigned*)tb)[i];
    int q = i >> 6;
    int d2 = (i & 63) * 2;
    s_tnT[d2 * PITCH + q] = cv.h[0];
    s_tnT[(d2 + 1) * PITCH + q] = cv.h[1];
  }
  __syncthreads();

  // c2q_un = P @ xq ; q2c_un = P @ T  (shared A fragments, batched B loads)
  v8f cacc[8], qacc[8];
#pragma unroll
  for (int j = 0; j < 8; ++j)
#pragma unroll
    for (int e = 0; e < 8; ++e) { cacc[j][e] = 0.f; qacc[j][e] = 0.f; }
#pragma unroll
  for (int kk = 0; kk < 4; ++kk) {
    v16bf a = frag_a_rowmajor(s_p, m0, kk * 32, lane);
    {
      v16bf bq[8];
#pragma unroll
      for (int j = 0; j < 8; ++j) bq[j] = frag_b_from_rows(s_xqT, j * 16, kk * 32, lane);
#pragma unroll
      for (int j = 0; j < 8; ++j) cacc[j] = wmma_bf16(a, bq[j], cacc[j]);
    }
    {
      v16bf bt[8];
#pragma unroll
      for (int j = 0; j < 8; ++j) bt[j] = frag_b_from_rows(s_tnT, j * 16, kk * 32, lane);
#pragma unroll
      for (int j = 0; j < 8; ++j) qacc[j] = wmma_bf16(a, bt[j], qacc[j]);
    }
  }

  // fused normalize + concat output: [ (xc done in k1), c2q, xc*c2q, xc*q2c ]
  {
    const int hi = lane >> 4, nl = lane & 15;
#pragma unroll
    for (int j = 0; j < 8; ++j) {
#pragma unroll
      for (int r = 0; r < 8; ++r) {
        const int mrow = m0 + r + 8 * hi;
        const int d = j * 16 + nl;
        const int c = c0 + mrow;
        const float inv = 1.0f / s_rowsum[mrow];
        const float c2q = cacc[j][r] * inv;
        const float q2c = qacc[j][r] * inv;
        const float xcv = xcb[(size_t)c * Dn + d];
        float* ob = out + ((size_t)b * LCn + c) * (4 * Dn);
        ob[Dn + d] = c2q;
        ob[2 * Dn + d] = xcv * c2q;
        ob[3 * Dn + d] = xcv * q2c;
      }
    }
  }
}

extern "C" void kernel_launch(void* const* d_in, const int* in_sizes, int n_in,
                              void* d_out, int out_size, void* d_ws, size_t ws_size,
                              hipStream_t stream) {
  const float* xc   = (const float*)d_in[0];
  const float* xq   = (const float*)d_in[1];
  const int*   clen = (const int*)d_in[2];
  const int*   qlen = (const int*)d_in[3];
  const float* W0   = (const float*)d_in[4];
  const float* W1   = (const float*)d_in[5];
  const float* W2   = (const float*)d_in[6];
  const float* bias = (const float*)d_in[7];
  float* out = (float*)d_out;
  __bf16* Tws = (__bf16*)d_ws;   // B*LQ*D bf16 = 2 MB

  k1_col_softmax_T<<<Bn * 2, 256, 0, stream>>>(xc, xq, clen, W0, W1, W2, bias, Tws, out);
  k2_row_softmax_out<<<Bn * 8, 256, 0, stream>>>(xc, xq, qlen, W0, W1, W2, bias, Tws, out);
}
